// S2VT_89575837925552
// MI455X (gfx1250) — compile-verified, matrix operand order checked by reading, not run on hardware
//
#include <hip/hip_runtime.h>

// ---------------------------------------------------------------------------
// S2VT (video captioning) for MI455X / gfx1250, wave32, bf16 WMMA w/ f32 accum
// Round 3: 2x-unrolled ping-pong K loop (no buffer-rotation register copies;
// the previous version spent ~40 v_dual_mov_b32 per iteration rotating frags).
// Requires K % 64 == 0 (true for all uses here: K in {4096, 1024}).
// ---------------------------------------------------------------------------

typedef __bf16 bf16;
typedef bf16  v16bf __attribute__((ext_vector_type(16)));
typedef bf16  v8bf  __attribute__((ext_vector_type(8)));
typedef float v8f   __attribute__((ext_vector_type(8)));

#define K_B   128
#define K_TV  80
#define K_F   4096
#define K_H   1024
#define K_V   12000
#define K_L   30
#define K_T   109   // TV + L - 1
#define K_TL  29    // L - 1

// --- WMMA fragment load: 16 bf16 per lane as two contiguous 16B chunks.
// 16-bit A layout (ISA 7.12.2): lanes 0-15 hold K = {kk..kk+7, kk+16..kk+23},
// lanes 16-31 hold K = {kk+8..kk+15, kk+24..kk+31}. B^T rows load identically.
__device__ __forceinline__ v16bf load_frag(const bf16* __restrict__ row,
                                           int kk, int grp) {
  union { v16bf v; v8bf h[2]; } u;
  const v8bf* p = reinterpret_cast<const v8bf*>(row + kk + (grp << 3));
  u.h[0] = p[0];   // kk + grp*8 .. +7
  u.h[1] = p[2];   // kk + grp*8 + 16 .. +23
  return u.v;
}

struct FragGroup {
  v16bf a;
  v16bf b[4];
};

__device__ __forceinline__ void load_group(FragGroup& f,
                                           const bf16* __restrict__ arow,
                                           const bf16* __restrict__ const* brow,
                                           int kk, int grp) {
  f.a = load_frag(arow, kk, grp);
#pragma unroll
  for (int j = 0; j < 4; ++j) f.b[j] = load_frag(brow[j], kk, grp);
}

__device__ __forceinline__ float sigmoidf_fast(float x) {
  return 1.0f / (1.0f + __expf(-x));
}

// ---------------------------------------------------------------------------
// Generic bf16 GEMM:  C[M,N] = A[M,K] @ Bt[N,K]^T  (+bias / +=)
// One wave computes a 16x64 strip (4 WMMA accumulators sharing one A frag).
// Ping-pong double buffering, unrolled by 2 so no register rotation is needed.
// Row remap for output: orow = (r / rin) * rout + roff + (r % rin)
//   (pass rin=M, rout=0, roff=0 for identity).
// ---------------------------------------------------------------------------
#define MODE_F32_BIAS  0
#define MODE_BF16_BIAS 1
#define MODE_F32_ACC   2

template <int MODE>
__global__ __launch_bounds__(128) void gemm_bf16_wmma(
    const bf16* __restrict__ A, const bf16* __restrict__ Bt,
    const float* __restrict__ bias, void* __restrict__ Cout,
    int M, int N, int K, int rin, int rout, int roff) {
  const int lane = threadIdx.x & 31;
  const int wave = blockIdx.x * 4 + (threadIdx.x >> 5);
  const int nstrips = (N + 63) >> 6;
  const int total = (M >> 4) * nstrips;
  if (wave >= total) return;                 // wave-uniform
  const int mt = wave / nstrips;
  const int ns = wave - mt * nstrips;
  const int m0 = mt << 4, n0 = ns << 6;
  const int grp = lane >> 4, lr = lane & 15;

  const bf16* __restrict__ arow = A + (size_t)(m0 + lr) * K;
  const bf16* __restrict__ brow[4];
  bool jok[4];
#pragma unroll
  for (int j = 0; j < 4; ++j) {
    const int nj = n0 + (j << 4);
    jok[j]  = (nj < N);                      // uniform; N always mult of 16
    brow[j] = Bt + (size_t)((jok[j] ? nj : 0) + lr) * K;
  }

  v8f acc[4] = {};
  FragGroup f0, f1;
  load_group(f0, arow, brow, 0, grp);        // preload kk = 0

  for (int kk = 0; kk < K - 64; kk += 64) {
    load_group(f1, arow, brow, kk + 32, grp);
#pragma unroll
    for (int j = 0; j < 4; ++j)
      if (jok[j])
        acc[j] = __builtin_amdgcn_wmma_f32_16x16x32_bf16(
            false, f0.a, false, f0.b[j], (short)0, acc[j], false, false);
    load_group(f0, arow, brow, kk + 64, grp);
#pragma unroll
    for (int j = 0; j < 4; ++j)
      if (jok[j])
        acc[j] = __builtin_amdgcn_wmma_f32_16x16x32_bf16(
            false, f1.a, false, f1.b[j], (short)0, acc[j], false, false);
  }
  // final 64: buf0 holds K-64; load K-32 into buf1
  load_group(f1, arow, brow, K - 32, grp);
#pragma unroll
  for (int j = 0; j < 4; ++j)
    if (jok[j])
      acc[j] = __builtin_amdgcn_wmma_f32_16x16x32_bf16(
          false, f0.a, false, f0.b[j], (short)0, acc[j], false, false);
#pragma unroll
  for (int j = 0; j < 4; ++j)
    if (jok[j])
      acc[j] = __builtin_amdgcn_wmma_f32_16x16x32_bf16(
          false, f1.a, false, f1.b[j], (short)0, acc[j], false, false);

#pragma unroll
  for (int j = 0; j < 4; ++j) {
    const int nj = n0 + (j << 4);
    if (!jok[j]) continue;
    const int col = nj + lr;
    const float bv = (MODE != MODE_F32_ACC) ? bias[col] : 0.0f;
#pragma unroll
    for (int v = 0; v < 8; ++v) {
      const int row  = m0 + (grp << 3) + v;
      const int orow = (row / rin) * rout + roff + (row % rin);
      const size_t oidx = (size_t)orow * N + col;
      if (MODE == MODE_F32_BIAS) {
        ((float*)Cout)[oidx] = acc[j][v] + bv;
      } else if (MODE == MODE_BF16_BIAS) {
        ((bf16*)Cout)[oidx] = (bf16)(acc[j][v] + bv);
      } else {
        ((float*)Cout)[oidx] += acc[j][v];
      }
    }
  }
}

// ---------------------------------------------------------------------------
// Fused LSTM step:  gates = Xpre[:,t,:] + h_in @ Whh^T ; elementwise update.
// Grid: 512 waves (8 M-tiles x 64 hidden slices). Each wave holds the FOUR
// gate tiles (i,f,g,o) for its 16-column hidden slice in 4 WMMA accumulators,
// sharing one A (h) fragment per K step; same ping-pong K loop as the GEMM.
// h double-buffered across steps (h_in != h_out).
// tail_mode==0: write h to seq_out[(b*T+t)*H]   (LSTM1, full sequence)
// tail_mode==1: write h to seq_out[(b*29+(t-80))*H] only when t>=80 (LSTM2)
// ---------------------------------------------------------------------------
__global__ __launch_bounds__(128) void lstm_step_wmma(
    const float* __restrict__ Xpre, const bf16* __restrict__ Whh,
    const bf16* __restrict__ h_in, bf16* __restrict__ h_out,
    float* __restrict__ c, bf16* __restrict__ seq_out,
    int t, int tail_mode) {
  const int lane = threadIdx.x & 31;
  const int wave = blockIdx.x * 4 + (threadIdx.x >> 5);  // 0..511
  const int mt = wave >> 6;        // 0..7   batch tile
  const int ns = wave & 63;        // 0..63  hidden slice
  const int m0 = mt << 4, n0 = ns << 4;
  const int grp = lane >> 4, lr = lane & 15;

  // init accumulators from precomputed input projection (already has biases)
  v8f acc[4];
#pragma unroll
  for (int g = 0; g < 4; ++g)
#pragma unroll
    for (int v = 0; v < 8; ++v) {
      const int row = m0 + (grp << 3) + v;
      acc[g][v] = Xpre[((size_t)row * K_T + t) * (4 * K_H) + g * K_H + n0 + lr];
    }

  const bf16* __restrict__ arow = h_in + (size_t)(m0 + lr) * K_H;
  const bf16* __restrict__ brow[4];
#pragma unroll
  for (int g = 0; g < 4; ++g)
    brow[g] = Whh + (size_t)(g * K_H + n0 + lr) * K_H;

  FragGroup f0, f1;
  load_group(f0, arow, brow, 0, grp);

  for (int kk = 0; kk < K_H - 64; kk += 64) {
    load_group(f1, arow, brow, kk + 32, grp);
#pragma unroll
    for (int g = 0; g < 4; ++g)
      acc[g] = __builtin_amdgcn_wmma_f32_16x16x32_bf16(
          false, f0.a, false, f0.b[g], (short)0, acc[g], false, false);
    load_group(f0, arow, brow, kk + 64, grp);
#pragma unroll
    for (int g = 0; g < 4; ++g)
      acc[g] = __builtin_amdgcn_wmma_f32_16x16x32_bf16(
          false, f1.a, false, f1.b[g], (short)0, acc[g], false, false);
  }
  load_group(f1, arow, brow, K_H - 32, grp);
#pragma unroll
  for (int g = 0; g < 4; ++g)
    acc[g] = __builtin_amdgcn_wmma_f32_16x16x32_bf16(
        false, f0.a, false, f0.b[g], (short)0, acc[g], false, false);
#pragma unroll
  for (int g = 0; g < 4; ++g)
    acc[g] = __builtin_amdgcn_wmma_f32_16x16x32_bf16(
        false, f1.a, false, f1.b[g], (short)0, acc[g], false, false);

#pragma unroll
  for (int v = 0; v < 8; ++v) {
    const int row = m0 + (grp << 3) + v;
    const int col = n0 + lr;
    const float xi = sigmoidf_fast(acc[0][v]);
    const float xf = sigmoidf_fast(acc[1][v]);
    const float xg = tanhf(acc[2][v]);
    const float xo = sigmoidf_fast(acc[3][v]);
    const size_t hc = (size_t)row * K_H + col;
    const float cn = xf * c[hc] + xi * xg;
    c[hc] = cn;
    const float hv = xo * tanhf(cn);
    const bf16 hb = (bf16)hv;
    h_out[hc] = hb;
    if (!tail_mode) {
      seq_out[((size_t)row * K_T + t) * K_H + col] = hb;
    } else if (t >= K_TV) {
      seq_out[((size_t)row * K_TL + (t - K_TV)) * K_H + col] = hb;
    }
  }
}

// ---------------------------------------------------------------------------
// Small helper kernels
// ---------------------------------------------------------------------------
__global__ __launch_bounds__(256) void k_f32_to_bf16(
    const float* __restrict__ s, bf16* __restrict__ d, size_t n) {
  for (size_t i = blockIdx.x * 256ull + threadIdx.x; i < n;
       i += (size_t)gridDim.x * 256ull)
    d[i] = (bf16)s[i];
}

// split-column convert: d[r*cols+c] = s[r*sld + soff + c]
__global__ __launch_bounds__(256) void k_convert_strided(
    const float* __restrict__ s, bf16* __restrict__ d,
    int rows, int cols, int sld, int soff) {
  size_t n = (size_t)rows * cols;
  for (size_t i = blockIdx.x * 256ull + threadIdx.x; i < n;
       i += (size_t)gridDim.x * 256ull) {
    int r = (int)(i / cols), cc = (int)(i % cols);
    d[i] = (bf16)s[(size_t)r * sld + soff + cc];
  }
}

// cap_bf[(b*29+tt)*H + h] = bf16(emb[captions[b*30+tt]*H + h])
__global__ __launch_bounds__(256) void k_gather_cap(
    const int* __restrict__ cap, const float* __restrict__ emb,
    bf16* __restrict__ out) {
  size_t n = (size_t)K_B * K_TL * K_H;
  for (size_t i = blockIdx.x * 256ull + threadIdx.x; i < n;
       i += (size_t)gridDim.x * 256ull) {
    int hid = (int)(i % K_H);
    int r   = (int)(i / K_H);       // b*29+tt
    int tt  = r % K_TL, b = r / K_TL;
    int tok = cap[b * K_L + tt];
    out[i] = (bf16)emb[(size_t)tok * K_H + hid];
  }
}

// X1 padded steps (t in [80,109)) get x = b1 exactly (zero feats)
__global__ __launch_bounds__(256) void k_fill_pad(
    bf16* __restrict__ X1, const float* __restrict__ b1) {
  size_t n = (size_t)K_B * K_TL * K_H;
  for (size_t i = blockIdx.x * 256ull + threadIdx.x; i < n;
       i += (size_t)gridDim.x * 256ull) {
    int hid = (int)(i % K_H);
    int r   = (int)(i / K_H);
    int t   = r % K_TL, b = r / K_TL;
    X1[((size_t)b * K_T + K_TV + t) * K_H + hid] = (bf16)b1[hid];
  }
}

__global__ __launch_bounds__(256) void k_add_bias(
    const float* __restrict__ a, const float* __restrict__ b,
    float* __restrict__ o, int n) {
  int i = blockIdx.x * 256 + threadIdx.x;
  if (i < n) o[i] = a[i] + b[i];
}

// ---------------------------------------------------------------------------
// Host orchestration
// ---------------------------------------------------------------------------
extern "C" void kernel_launch(void* const* d_in, const int* in_sizes, int n_in,
                              void* d_out, int out_size, void* d_ws,
                              size_t ws_size, hipStream_t stream) {
  (void)in_sizes; (void)n_in; (void)out_size; (void)ws_size;
  const float* video = (const float*)d_in[0];   // (128,80,4096)
  const int*   caps  = (const int*)  d_in[1];   // (128,30)
  const float* W1    = (const float*)d_in[2];   // (1024,4096)
  const float* b1    = (const float*)d_in[3];   // (1024,)
  const float* emb   = (const float*)d_in[4];   // (12000,1024)
  const float* Wih1  = (const float*)d_in[5];   // (4096,1024)
  const float* Whh1  = (const float*)d_in[6];   // (4096,1024)
  const float* bih1  = (const float*)d_in[7];
  const float* bhh1  = (const float*)d_in[8];
  const float* Wih2  = (const float*)d_in[9];   // (4096,2048)
  const float* Whh2  = (const float*)d_in[10];  // (4096,1024)
  const float* bih2  = (const float*)d_in[11];
  const float* bhh2  = (const float*)d_in[12];
  const float* W2    = (const float*)d_in[13];  // (12000,1024)
  const float* b2    = (const float*)d_in[14];  // (12000,)
  float* logits = (float*)d_out;                // (128,29,12000) f32

  // ---- workspace bump allocator (256B aligned) ----
  size_t off = 0;
  auto alloc = [&](size_t bytes) -> void* {
    off = (off + 255) & ~(size_t)255;
    void* p = (char*)d_ws + off;
    off += bytes;
    return p;
  };
  const size_t BT  = (size_t)K_B * K_T;        // 13952
  const size_t BTV = (size_t)K_B * K_TV;       // 10240
  const size_t BTL = (size_t)K_B * K_TL;       // 3712

  bf16* feats_bf = (bf16*)alloc(BTV * K_F * 2);           // 84 MB
  bf16* W1_bf    = (bf16*)alloc((size_t)K_H * K_F * 2);
  bf16* Wih1_bf  = (bf16*)alloc((size_t)4 * K_H * K_H * 2);
  bf16* Whh1_bf  = (bf16*)alloc((size_t)4 * K_H * K_H * 2);
  bf16* Wih2a_bf = (bf16*)alloc((size_t)4 * K_H * K_H * 2);
  bf16* Wih2b_bf = (bf16*)alloc((size_t)4 * K_H * K_H * 2);
  bf16* Whh2_bf  = (bf16*)alloc((size_t)4 * K_H * K_H * 2);
  bf16* W2_bf    = (bf16*)alloc((size_t)K_V * K_H * 2);
  bf16* X1_bf    = (bf16*)alloc(BT * K_H * 2);            // 28.6 MB
  float* Xpre    = (float*)alloc(BT * 4 * K_H * 4);       // 229 MB (reused!)
  bf16* l1out_bf = (bf16*)alloc(BT * K_H * 2);
  bf16* cap_bf   = (bf16*)alloc(BTL * K_H * 2);
  bf16* tail_bf  = (bf16*)alloc(BTL * K_H * 2);
  bf16* h1[2] = {(bf16*)alloc((size_t)K_B * K_H * 2),
                 (bf16*)alloc((size_t)K_B * K_H * 2)};
  bf16* h2[2] = {(bf16*)alloc((size_t)K_B * K_H * 2),
                 (bf16*)alloc((size_t)K_B * K_H * 2)};
  float* c1    = (float*)alloc((size_t)K_B * K_H * 4);
  float* c2    = (float*)alloc((size_t)K_B * K_H * 4);
  float* bsum1 = (float*)alloc(4 * K_H * 4);
  float* bsum2 = (float*)alloc(4 * K_H * 4);

  auto ew_blocks = [](size_t n) {
    size_t b = (n + 255) / 256;
    return (int)(b > 4096 ? 4096 : b);
  };
  auto gemm_blocks = [](int M, int N) {
    long long waves = (long long)(M >> 4) * ((N + 63) >> 6);
    return (int)((waves + 3) >> 2);
  };

  // ---- conversions / setup ----
  k_f32_to_bf16<<<ew_blocks(BTV * K_F), 256, 0, stream>>>(video, feats_bf,
                                                          BTV * K_F);
  k_f32_to_bf16<<<ew_blocks((size_t)K_H * K_F), 256, 0, stream>>>(
      W1, W1_bf, (size_t)K_H * K_F);
  k_f32_to_bf16<<<ew_blocks((size_t)4 * K_H * K_H), 256, 0, stream>>>(
      Wih1, Wih1_bf, (size_t)4 * K_H * K_H);
  k_f32_to_bf16<<<ew_blocks((size_t)4 * K_H * K_H), 256, 0, stream>>>(
      Whh1, Whh1_bf, (size_t)4 * K_H * K_H);
  k_f32_to_bf16<<<ew_blocks((size_t)4 * K_H * K_H), 256, 0, stream>>>(
      Whh2, Whh2_bf, (size_t)4 * K_H * K_H);
  k_f32_to_bf16<<<ew_blocks((size_t)K_V * K_H), 256, 0, stream>>>(
      W2, W2_bf, (size_t)K_V * K_H);
  k_convert_strided<<<ew_blocks((size_t)4 * K_H * K_H), 256, 0, stream>>>(
      Wih2, Wih2a_bf, 4 * K_H, K_H, 2 * K_H, 0);
  k_convert_strided<<<ew_blocks((size_t)4 * K_H * K_H), 256, 0, stream>>>(
      Wih2, Wih2b_bf, 4 * K_H, K_H, 2 * K_H, K_H);
  k_add_bias<<<16, 256, 0, stream>>>(bih1, bhh1, bsum1, 4 * K_H);
  k_add_bias<<<16, 256, 0, stream>>>(bih2, bhh2, bsum2, 4 * K_H);
  k_gather_cap<<<ew_blocks(BTL * K_H), 256, 0, stream>>>(caps, emb, cap_bf);
  k_fill_pad<<<ew_blocks(BTL * K_H), 256, 0, stream>>>(X1_bf, b1);

  hipMemsetAsync(h1[0], 0, (size_t)K_B * K_H * 2, stream);
  hipMemsetAsync(h2[0], 0, (size_t)K_B * K_H * 2, stream);
  hipMemsetAsync(c1, 0, (size_t)K_B * K_H * 4, stream);
  hipMemsetAsync(c2, 0, (size_t)K_B * K_H * 4, stream);

  // ---- X1 = feats @ W1^T + b1  (bf16, rows remapped b*80+t -> b*109+t) ----
  gemm_bf16_wmma<MODE_BF16_BIAS>
      <<<gemm_blocks((int)BTV, K_H), 128, 0, stream>>>(
          feats_bf, W1_bf, b1, X1_bf, (int)BTV, K_H, K_F, K_TV, K_T, 0);

  // ---- X1pre = X1 @ Wih1^T + (bih1+bhh1) : f32 ----
  gemm_bf16_wmma<MODE_F32_BIAS>
      <<<gemm_blocks((int)BT, 4 * K_H), 128, 0, stream>>>(
          X1_bf, Wih1_bf, bsum1, Xpre, (int)BT, 4 * K_H, K_H, (int)BT, 0, 0);

  // ---- LSTM1 sequential scan ----
  for (int t = 0; t < K_T; ++t) {
    lstm_step_wmma<<<128, 128, 0, stream>>>(
        Xpre, Whh1_bf, h1[t & 1], h1[(t + 1) & 1], c1, l1out_bf, t, 0);
  }

  // ---- X2pre = lstm1_out @ Wih2a^T + (bih2+bhh2); += cap @ Wih2b^T ----
  gemm_bf16_wmma<MODE_F32_BIAS>
      <<<gemm_blocks((int)BT, 4 * K_H), 128, 0, stream>>>(
          l1out_bf, Wih2a_bf, bsum2, Xpre, (int)BT, 4 * K_H, K_H, (int)BT, 0, 0);
  gemm_bf16_wmma<MODE_F32_ACC>
      <<<gemm_blocks((int)BTL, 4 * K_H), 128, 0, stream>>>(
          cap_bf, Wih2b_bf, nullptr, Xpre, (int)BTL, 4 * K_H, K_H,
          K_TL, K_T, K_TV);

  // ---- LSTM2 sequential scan (keep only tail t>=80) ----
  for (int t = 0; t < K_T; ++t) {
    lstm_step_wmma<<<128, 128, 0, stream>>>(
        Xpre, Whh2_bf, h2[t & 1], h2[(t + 1) & 1], c2, tail_bf, t, 1);
  }

  // ---- logits = tail @ W2^T + b2 : (3712, 12000) f32 straight to d_out ----
  gemm_bf16_wmma<MODE_F32_BIAS>
      <<<gemm_blocks((int)BTL, K_V), 128, 0, stream>>>(
          tail_bf, W2_bf, b2, logits, (int)BTL, K_V, K_H, (int)BTL, 0, 0);
}
